// TransformerEncoder_40492951667372
// MI455X (gfx1250) — matrix operand order repaired and checked by reading, not compiled, not run on hardware
//
#include <hip/hip_runtime.h>
#include <hip/hip_bf16.h>

// ---------------- problem constants ----------------
constexpr int cB   = 8;
constexpr int cL   = 1024;
constexpr int cD   = 64;
constexpr int cH   = 16;
constexpr int cDH  = 4;          // head dim -> matches V_WMMA_F32_16X16X4_F32
constexpr int cDFF = 256;
constexpr int cT   = cB * cL;    // 8192 tokens
constexpr float cEPS = 1e-5f;

typedef __attribute__((ext_vector_type(2)))  float          v2f;
typedef __attribute__((ext_vector_type(4)))  float          v4f;
typedef __attribute__((ext_vector_type(8)))  float          v8f;
typedef __attribute__((ext_vector_type(16))) __bf16         v16bf;
typedef __attribute__((ext_vector_type(4)))  unsigned short u16x4;
typedef __attribute__((ext_vector_type(8)))  unsigned short u16x8;
typedef __attribute__((ext_vector_type(8)))  unsigned int   u32x8;

struct U16x16 { u16x8 lo, hi; };   // 32B, bit-castable to v16bf

// ---------------- bf16 helpers: native hardware convert ----------------
__device__ __forceinline__ unsigned short f2bfu(float f) {
  __bf16 h = (__bf16)f;                       // native v_cvt on gfx1250
  return __builtin_bit_cast(unsigned short, h);
}
__device__ __forceinline__ unsigned pk_bf16(float a, float b) {
  return (unsigned)f2bfu(a) | ((unsigned)f2bfu(b) << 16);
}

// ---------------- f32 -> bf16 convert (x4 vectorized) ----------------
__global__ void k_cvt_bf16(const float* __restrict__ in,
                           unsigned short* __restrict__ out, int n) {
  int i = (blockIdx.x * blockDim.x + threadIdx.x) * 4;
  if (i >= n) return;
  v4f x = *(const v4f*)(in + i);
  u16x4 o;
  o[0] = f2bfu(x[0]); o[1] = f2bfu(x[1]);
  o[2] = f2bfu(x[2]); o[3] = f2bfu(x[3]);
  *(u16x4*)(out + i) = o;
}

// ---------------- GEMM: C[t,n] = act(A[t,:]·W[n,:] + bias[n]) (+res) ----------
// A: [cT x K] bf16, W: [Nout x K] bf16 (both row-major).
// One wave computes a 16x16 tile via v_wmma_f32_16x16x32_bf16.
// Fragments loaded as 2x16B vectors (CDNA5 bf16 A/B K-interleave is chunk-contiguous).
// block = 128 (4 waves); grid = (cT/16, Nout/64).
// TRB: store bf16 output transposed as outB[n*cT + t] (for attention V).
template<int K, bool RELU, bool RES, bool TRB>
__global__ void k_gemm(const unsigned short* __restrict__ A,
                       const unsigned short* __restrict__ W,
                       const float* __restrict__ bias,
                       const float* __restrict__ res,
                       float* __restrict__ outF,
                       unsigned short* __restrict__ outB,
                       int Nout) {
  const int lane = threadIdx.x & 31;
  const int wave = threadIdx.x >> 5;
  const int half = lane >> 4;       // K-half selector (wave32 layout)
  const int lr   = lane & 15;       // row (A) / col (B) within tile
  const int t0   = blockIdx.x * 16;
  const int n0   = (blockIdx.y * 4 + wave) * 16;
  const int row  = t0 + lr;

  v8f c;
#pragma unroll
  for (int i = 0; i < 8; ++i) c[i] = 0.f;

#pragma unroll
  for (int kb = 0; kb < K; kb += 32) {
    // A 16x32 bf16 layout: lanes0-15 K in {0-7,16-23}, lanes16-31 {8-15,24-31}
    U16x16 ap, wp;
    ap.lo = *(const u16x8*)(A + row * K + kb + half * 8);
    ap.hi = *(const u16x8*)(A + row * K + kb + 16 + half * 8);
    // B 32x16 bf16 layout: lane = column n, lanes0-15 K in {0-7,16-23}, lanes16-31 {8-15,24-31}
    wp.lo = *(const u16x8*)(W + (n0 + lr) * K + kb + half * 8);
    wp.hi = *(const u16x8*)(W + (n0 + lr) * K + kb + 16 + half * 8);
    v16bf a = __builtin_bit_cast(v16bf, ap);
    v16bf w = __builtin_bit_cast(v16bf, wp);
    c = __builtin_amdgcn_wmma_f32_16x16x32_bf16(false, a, false, w,
                                                (short)0, c, false, false);
  }

#pragma unroll
  for (int i = 0; i < 8; ++i) {
    int m = i + 8 * half;        // C-layout: VGPR i -> row i (half0) / 8+i (half1)
    int t = t0 + m;
    int n = n0 + lr;
    float v = c[i] + bias[n];
    if (RELU) v = fmaxf(v, 0.f);
    if (RES)  v += res[t * Nout + n];
    if (outF) outF[t * Nout + n] = v;
    if (outB) outB[TRB ? (n * cT + t) : (t * Nout + n)] = f2bfu(v);
  }
}

// ---------------- BatchNorm stats: 2-stage, coalesced, deterministic --------
// Stage 1: 64 blocks x 256 threads; block j covers tokens [j*128, j*128+128).
// tid = c + 64*sub -> consecutive lanes read consecutive channels (coalesced).
__global__ void k_bn_part(const float* __restrict__ X, float* __restrict__ part) {
  const int j   = blockIdx.x;
  const int tid = threadIdx.x;
  const int c   = tid & 63;
  const int sub = tid >> 6;          // 0..3
  const int tbase = j * 128;
  float s = 0.f, s2 = 0.f;
  for (int tt = sub; tt < 128; tt += 4) {
    float v = X[(tbase + tt) * cD + c];
    s += v; s2 += v * v;
  }
  __shared__ float sh1[256], sh2[256];
  sh1[tid] = s; sh2[tid] = s2;
  __syncthreads();
  if (sub == 0) {
    float a  = sh1[c] + sh1[64 + c] + sh1[128 + c] + sh1[192 + c];
    float b2 = sh2[c] + sh2[64 + c] + sh2[128 + c] + sh2[192 + c];
    part[j * cD + c]            = a;
    part[64 * cD + j * cD + c]  = b2;
  }
}

// Stage 2: one tiny block folds 64 partials per channel into mean/var.
__global__ void k_bn_final(const float* __restrict__ part,
                           float* __restrict__ stats) {
  const int c = threadIdx.x;         // 64 threads
  float s = 0.f, s2 = 0.f;
  for (int j = 0; j < 64; ++j) {
    s  += part[j * cD + c];
    s2 += part[64 * cD + j * cD + c];
  }
  float mean = s / (float)cT;
  float var  = s2 / (float)cT - mean * mean;   // biased, as torch
  stats[c] = mean; stats[cD + c] = var;
}

// BN apply (x4 vectorized), fused with bf16 re-encode of enc.
__global__ void k_bn_apply(const float* __restrict__ X,
                           const float* __restrict__ stats,
                           const float* __restrict__ g,
                           const float* __restrict__ be,
                           float* __restrict__ encF,
                           unsigned short* __restrict__ encB) {
  int i = (blockIdx.x * 256 + threadIdx.x) * 4;
  if (i >= cT * cD) return;
  v4f x = *(const v4f*)(X + i);
  int c0 = i & (cD - 1);             // i % 64, multiple of 4 -> c0..c0+3 in range
  v4f y;
  u16x4 yb;
#pragma unroll
  for (int k = 0; k < 4; ++k) {
    int c = c0 + k;
    float v = g[c] * (x[k] - stats[c]) * rsqrtf(stats[cD + c] + cEPS) + be[c];
    y[k] = v;
    yb[k] = f2bfu(v);
  }
  *(v4f*)(encF + i)   = y;
  *(u16x4*)(encB + i) = yb;
}

// ---------------- Flash attention, head dim 4, fully transposed -------------
// grid = (L/128, H, B); block = 256 (8 waves). Each wave: 16 query columns.
// S^T = K·Q^T via v_wmma_f32_16x16x4_f32 (K = DH = 4, exact): lane = l, so
// softmax rows reduce in registers + one xor16 shuffle; running max/sum/alpha
// are per-lane scalars. O^T = V^T · P^T via v_wmma_f32_16x16x32_bf16 — the
// exp'd S^T values become the B-operand after exchanging the middle m-halves
// as 4 packed-bf16 words. No LDS round-trip at all.
__global__ void k_attn(const float* __restrict__ Q,
                       const float* __restrict__ Kf,
                       const unsigned short* __restrict__ Vt,  // [cD][cT] bf16
                       const float* __restrict__ enc,
                       float* __restrict__ out) {
  const int lane = threadIdx.x & 31;
  const int wave = threadIdx.x >> 5;
  const int lr   = lane & 15;
  const int half = lane >> 4;
  const int b    = blockIdx.z;
  const int h4   = blockIdx.y * cDH;
  const int l0   = blockIdx.x * 128 + wave * 16;
  const int tb   = b * cL;

  // Q as B-operand of 4x16 (col = l = lr, K pair per half); 1/sqrt(DH) folded in
  const int koff = half * 2;
  v2f qb = *(const v2f*)(&Q[(tb + l0 + lr) * cD + h4 + koff]);
  qb.x *= 0.5f; qb.y *= 0.5f;

  // V^T A-fragment base (row n = lr, only n < DH valid; rest zero-padded)
  const unsigned short* vrow = Vt + (h4 + (lr & 3)) * cT + tb;
  u16x8 zv;
#pragma unroll
  for (int j = 0; j < 8; ++j) zv[j] = 0;

  float rmax = -3.0e38f, rsum = 0.f;
  v8f acc;
#pragma unroll
  for (int i = 0; i < 8; ++i) acc[i] = 0.f;

  for (int m0 = 0; m0 < cL; m0 += 32) {
    if (m0 + 32 < cL)
      __builtin_prefetch(&Kf[(tb + m0 + 32 + lr) * cD + h4], 0, 0);

    // ---- S^T tiles: A = K fragment (16m x 4), B = Q^T ----
    v2f ka0 = *(const v2f*)(&Kf[(tb + m0 + lr) * cD + h4 + koff]);
    v2f ka1 = *(const v2f*)(&Kf[(tb + m0 + 16 + lr) * cD + h4 + koff]);
    v8f z;
#pragma unroll
    for (int i = 0; i < 8; ++i) z[i] = 0.f;
    v8f s0 = __builtin_amdgcn_wmma_f32_16x16x4_f32(false, ka0, false, qb,
                                                   (short)0, z, false, false);
    v8f s1 = __builtin_amdgcn_wmma_f32_16x16x4_f32(false, ka1, false, qb,
                                                   (short)0, z, false, false);
    // lane (half,lr) holds column l = lr:  s0 -> m = i + 8*half,
    //                                      s1 -> m = 16 + i + 8*half

    // ---- online softmax: register reduction + one xor16 combine ----
    float tm = fmaxf(s0[0], s1[0]);
#pragma unroll
    for (int i = 1; i < 8; ++i) tm = fmaxf(tm, fmaxf(s0[i], s1[i]));
    tm = fmaxf(tm, __shfl_xor(tm, 16, 32));
    float nm = fmaxf(rmax, tm);
    float al = __expf(rmax - nm);
    rmax = nm;

    float p0[8], p1[8], ts = 0.f;
#pragma unroll
    for (int i = 0; i < 8; ++i) {
      p0[i] = __expf(s0[i] - nm);
      p1[i] = __expf(s1[i] - nm);
      ts += p0[i] + p1[i];
    }
    ts += __shfl_xor(ts, 16, 32);
    rsum = rsum * al + ts;
#pragma unroll
    for (int i = 0; i < 8; ++i) acc[i] *= al;   // per-lane alpha: no broadcast

    // ---- P^T B-fragment: convert to packed bf16, exchange middle m-halves
    //      across the lane pair with 4 word shuffles, bit-cast to v16bf.
    // half0 keeps p0 (m0-7) in words0-3, receives m8-15 into words4-7;
    // half1 receives m16-23 into words0-3, keeps p1 (m24-31) in words4-7.
    u32x8 pw;
#pragma unroll
    for (int w = 0; w < 4; ++w) {
      unsigned own  = half ? pk_bf16(p1[2*w], p1[2*w+1])
                           : pk_bf16(p0[2*w], p0[2*w+1]);
      unsigned send = half ? pk_bf16(p0[2*w], p0[2*w+1])
                           : pk_bf16(p1[2*w], p1[2*w+1]);
      unsigned recv = (unsigned)__shfl_xor((int)send, 16, 32);
      pw[w]     = half ? recv : own;
      pw[4 + w] = half ? own  : recv;
    }
    v16bf pb = __builtin_bit_cast(v16bf, pw);

    // ---- V^T A-fragment: row n = lr (n<4 valid), K-contiguous per half (m = 16*half + s) ----
    v16bf va;
    if (lr < cDH) {
      U16x16 vp;
      vp.lo = *(const u16x8*)(vrow + m0 + half * 16);       // m: {0-7|16-23}
      vp.hi = *(const u16x8*)(vrow + m0 + half * 16 + 8);   // m: {8-15|24-31}
      va = __builtin_bit_cast(v16bf, vp);
    } else {
      va = __builtin_bit_cast(v16bf, U16x16{zv, zv});
    }

    // ---- O^T accumulate: rows = d, cols = l ----
    acc = __builtin_amdgcn_wmma_f32_16x16x32_bf16(false, va, false, pb,
                                                  (short)0, acc, false, false);
  }

  // O^T C-layout: VGPR i -> row n = i + 8*half, col l = lr.
  // Valid d rows are 0..3 -> half 0, i < 4. rsum is per-lane (l = lr).
  if (half == 0) {
    float inv = 1.f / rsum;
#pragma unroll
    for (int i = 0; i < cDH; ++i) {
      int idx = (tb + l0 + lr) * cD + h4 + i;
      out[idx] = acc[i] * inv + enc[idx];   // fused residual
    }
  }
}

// ---------------- driver ----------------
extern "C" void kernel_launch(void* const* d_in, const int* in_sizes, int n_in,
                              void* d_out, int out_size, void* d_ws, size_t ws_size,
                              hipStream_t stream) {
  const float* x       = (const float*)d_in[0];
  const float* We      = (const float*)d_in[1];
  const float* b_embed = (const float*)d_in[2];
  const float* g0      = (const float*)d_in[3];
  const float* be0     = (const float*)d_in[4];
  const float* Wq      = (const float*)d_in[5];
  const float* bq      = (const float*)d_in[6];
  const float* Wk      = (const float*)d_in[7];
  const float* bk      = (const float*)d_in[8];
  const float* Wv      = (const float*)d_in[9];
  const float* bv      = (const float*)d_in[10];
  const float* g_attn  = (const float*)d_in[11];
  const float* be_attn = (const float*)d_in[12];
  const float* W1      = (const float*)d_in[13];
  const float* b1      = (const float*)d_in[14];
  const float* W2      = (const float*)d_in[15];
  const float* b2      = (const float*)d_in[16];
  const float* g_ff    = (const float*)d_in[17];
  const float* be_ff   = (const float*)d_in[18];

  char* ws = (char*)d_ws;
  size_t off = 0;
  auto carve = [&](size_t bytes) -> void* {
    void* p = ws + off;
    off += (bytes + 255) & ~(size_t)255;
    return p;
  };

  unsigned short* x_bf  = (unsigned short*)carve((size_t)cT * cD * 2);
  unsigned short* We_bf = (unsigned short*)carve((size_t)cD * cD * 2);
  unsigned short* Wq_bf = (unsigned short*)carve((size_t)cD * cD * 2);
  unsigned short* Wk_bf = (unsigned short*)carve((size_t)cD * cD * 2);
  unsigned short* Wv_bf = (unsigned short*)carve((size_t)cD * cD * 2);
  unsigned short* W1_bf = (unsigned short*)carve((size_t)cDFF * cD * 2);
  unsigned short* W2_bf = (unsigned short*)carve((size_t)cD * cDFF * 2);
  float*          pre   = (float*)carve((size_t)cT * cD * 4);   // pre-BN buffer
  float*          encF  = (float*)carve((size_t)cT * cD * 4);
  unsigned short* encB  = (unsigned short*)carve((size_t)cT * cD * 2);
  float*          Qf    = (float*)carve((size_t)cT * cD * 4);
  float*          Kfb   = (float*)carve((size_t)cT * cD * 4);
  unsigned short* Vt_bf = (unsigned short*)carve((size_t)cT * cD * 2); // [cD][cT]
  unsigned short* h1_bf = (unsigned short*)carve((size_t)cT * cDFF * 2);
  float*          part  = (float*)carve((size_t)2 * 64 * cD * 4);
  float*          stats = (float*)carve((size_t)2 * cD * 4);

  auto cvt = [&](const float* src, unsigned short* dst, int n) {
    k_cvt_bf16<<<(n / 4 + 255) / 256, 256, 0, stream>>>(src, dst, n);
  };
  cvt(x,  x_bf,  cT * cD);
  cvt(We, We_bf, cD * cD);
  cvt(Wq, Wq_bf, cD * cD);
  cvt(Wk, Wk_bf, cD * cD);
  cvt(Wv, Wv_bf, cD * cD);
  cvt(W1, W1_bf, cDFF * cD);
  cvt(W2, W2_bf, cD * cDFF);

  auto bn = [&](const float* X, const float* g, const float* be) {
    k_bn_part<<<64, 256, 0, stream>>>(X, part);
    k_bn_final<<<1, 64, 0, stream>>>(part, stats);
    k_bn_apply<<<(cT * cD / 4) / 256, 256, 0, stream>>>(X, stats, g, be,
                                                        encF, encB);
  };

  const dim3 gemmBlk(128);
  const dim3 g64(cT / 16, 1);     // Nout = 64
  const dim3 g256(cT / 16, 4);    // Nout = 256

  // input embedding + BN
  k_gemm<64, false, false, false><<<g64, gemmBlk, 0, stream>>>(
      x_bf, We_bf, b_embed, nullptr, pre, nullptr, cD);
  bn(pre, g0, be0);

  for (int s = 0; s < 3; ++s) {
    // QKV projections (ReLU); V is emitted bf16 + transposed for attention
    k_gemm<64, true, false, false><<<g64, gemmBlk, 0, stream>>>(
        encB, Wq_bf, bq, nullptr, Qf, nullptr, cD);
    k_gemm<64, true, false, false><<<g64, gemmBlk, 0, stream>>>(
        encB, Wk_bf, bk, nullptr, Kfb, nullptr, cD);
    k_gemm<64, true, false, true><<<g64, gemmBlk, 0, stream>>>(
        encB, Wv_bf, bv, nullptr, nullptr, Vt_bf, cD);

    // attention (fused residual) + BN
    k_attn<<<dim3(cL / 128, cH, cB), 256, 0, stream>>>(Qf, Kfb, Vt_bf, encF, pre);
    bn(pre, g_attn, be_attn);

    // feedforward: FF1 (relu, bf16 out) -> FF2 (relu + residual) + BN
    k_gemm<64, true, false, false><<<g256, gemmBlk, 0, stream>>>(
        encB, W1_bf, b1, nullptr, nullptr, h1_bf, cDFF);
    k_gemm<256, true, true, false><<<g64, gemmBlk, 0, stream>>>(
        h1_bf, W2_bf, b2, encF, pre, nullptr, cD);
    bn(pre, g_ff, be_ff);
  }

  hipMemcpyAsync(d_out, encF, (size_t)cT * cD * sizeof(float),
                 hipMemcpyDeviceToDevice, stream);
}
